// ConvTransE_80917183857180
// MI455X (gfx1250) — matrix operand order repaired and checked by reading, not compiled, not run on hardware
//
#include <hip/hip_runtime.h>
#include <hip/hip_bf16.h>

#define DIMV     512
#define CHANNELS 32
#define KDIM     (CHANNELS * DIMV)   // 16384
#define BATCH    2048
#define NTHREADS 256
#define KC       64                  // K-chunk staged in LDS per TDM transfer
#define BROWS    32                  // M rows per block
#define BUF_ELEM (BROWS * KC + DIMV * KC)        // A tile + B tile (bf16 elems)
#define BUF_B    (BUF_ELEM * 2)                  // bytes per LDS buffer (69632)

typedef __attribute__((ext_vector_type(16))) __bf16         v16bf;
typedef __attribute__((ext_vector_type(8)))  float          v8f;
typedef __attribute__((ext_vector_type(16))) unsigned short v16u;
typedef __attribute__((ext_vector_type(8)))  unsigned short v8u;
typedef __attribute__((ext_vector_type(4)))  unsigned int   u32x4;
typedef __attribute__((ext_vector_type(8)))  int            i32x8;
typedef __attribute__((ext_vector_type(4)))  int            i32x4;

#if defined(__has_builtin)
#if __has_builtin(__builtin_amdgcn_tensor_load_to_lds) && __has_builtin(__builtin_amdgcn_s_wait_tensorcnt)
#define USE_TDM 1
#endif
#endif

__device__ __forceinline__ unsigned short f2bf(float f) {
    unsigned int u = __builtin_bit_cast(unsigned int, f);
    unsigned int r = u + 0x7FFFu + ((u >> 16) & 1u);   // round-to-nearest-even
    return (unsigned short)(r >> 16);
}

#ifdef USE_TDM
// Build a TDM D# for a 2-D tile of 2-byte elements and issue TENSOR_LOAD_TO_LDS.
//   tile: tile_k elems (contiguous, dim0) x tile_rows rows; row stride = stride_elems.
// D# layout per CDNA5 ISA.
// 6-arg builtin variant (clang-23 / therock-10.0): (g0, g1, g2, g3, g_extra, cpol).
__device__ __forceinline__ void tdm_load_2d_bf16(uint32_t lds_off, const void* gptr,
                                                 uint32_t tile_k, uint32_t tile_rows,
                                                 uint32_t stride_elems) {
    const uint64_t ga = (uint64_t)(uintptr_t)gptr;
    u32x4 g0;
    g0[0] = 1u;                                            // count=1, user D#, no gather
    g0[1] = lds_off;                                       // lds_addr (bytes)
    g0[2] = (uint32_t)ga;                                  // global_addr[31:0]
    g0[3] = (uint32_t)((ga >> 32) & 0x1FFFFFFu) | (2u << 30); // addr[56:32] | type=2
    i32x8 g1;
    g1[0] = (int)(1u << 16);                               // wg_mask=0, data_size=1 (2B)
    g1[1] = (int)((stride_elems & 0xFFFFu) << 16);         // tensor_dim0[15:0]
    g1[2] = (int)(((stride_elems >> 16) & 0xFFFFu) |       // tensor_dim0[31:16]
                  ((tile_rows & 0xFFFFu) << 16));          // tensor_dim1[15:0]
    g1[3] = (int)((tile_k & 0xFFFFu) << 16);               // tensor_dim1[31:16]=0 | tile_dim0
    g1[4] = (int)(tile_rows & 0xFFFFu);                    // tile_dim1 | tile_dim2=0
    g1[5] = (int)stride_elems;                             // tensor_dim0_stride[31:0]
    g1[6] = 0;                                             // stride0[47:32] | stride1[15:0]
    g1[7] = 0;                                             // tensor_dim1_stride[47:16]
    const i32x4 z4 = {0, 0, 0, 0};                         // groups 2/3 unused (2-D tile)
    const i32x8 z8 = {0, 0, 0, 0, 0, 0, 0, 0};
    __builtin_amdgcn_tensor_load_to_lds(g0, g1, z4, z4, z8, 0);
}
#endif

// ---------------------------------------------------------------- kernel 1:
// proj_w f32 -> bf16 (vectorized: 4 elems/thread)
__global__ __launch_bounds__(NTHREADS) void cvt_w_kernel(
    const float* __restrict__ w, unsigned short* __restrict__ wb, int n4) {
    int i = blockIdx.x * NTHREADS + threadIdx.x;
    if (i < n4) {
        const float4 f = ((const float4*)w)[i];
        ushort4 o;
        o.x = f2bf(f.x); o.y = f2bf(f.y); o.z = f2bf(f.z); o.w = f2bf(f.w);
        ((ushort4*)wb)[i] = o;
    }
}

// ---------------------------------------------------------------- kernel 2:
// gather + 1D conv (SAME,k=3) on surviving crop [0,512) + bias + ReLU -> bf16 X
__global__ __launch_bounds__(NTHREADS) void conv_gather_kernel(
    const float* __restrict__ ent, const float* __restrict__ rel,
    const int* __restrict__ h, const int* __restrict__ r,
    const float* __restrict__ cw, const float* __restrict__ cb,
    unsigned short* __restrict__ X) {
    __shared__ float sin[DIMV + 2];      // [0]=pad, [1..512]=h_e, [513]=r_e[0]
    __shared__ float sw[CHANNELS * 3];
    __shared__ float sb[CHANNELS];

    const int b   = blockIdx.x;
    const int tid = threadIdx.x;
    const float* he = ent + (size_t)h[b] * DIMV;

    for (int j = tid; j < DIMV; j += NTHREADS) sin[j + 1] = he[j];
    if (tid == 0) { sin[0] = 0.0f; sin[DIMV + 1] = rel[(size_t)r[b] * DIMV]; }
    if (tid < CHANNELS * 3) sw[tid] = cw[tid];
    if (tid < CHANNELS)     sb[tid] = cb[tid];
    __syncthreads();

    unsigned short* xrow = X + (size_t)b * KDIM;
    for (int idx = tid; idx < KDIM; idx += NTHREADS) {
        const int c = idx >> 9;
        const int j = idx & (DIMV - 1);
        float v = sb[c] + sw[c * 3 + 0] * sin[j]
                        + sw[c * 3 + 1] * sin[j + 1]
                        + sw[c * 3 + 2] * sin[j + 2];
        v = v > 0.0f ? v : 0.0f;
        xrow[idx] = f2bf(v);
    }
}

// ---------------------------------------------------------------- kernel 3:
// Y = X(2048x16384 bf16) . Wbf(512x16384 bf16)^T + proj_b  via WMMA.
// Block: 32 M-rows x all 512 cols. Wave: 2 M-tiles x 4 N-tiles (8 acc tiles).
// K staged in LDS in KC=64 chunks, double-buffered, filled by the TDM.
__global__ __launch_bounds__(NTHREADS) void gemm_wmma_kernel(
    const unsigned short* __restrict__ X, const unsigned short* __restrict__ Wb,
    const float* __restrict__ pb, float* __restrict__ Y) {
    extern __shared__ char smem[];   // 2 buffers: [A: 32xKC][B: 512xKC] bf16

    const int lane = threadIdx.x & 31;
    const int wave = threadIdx.x >> 5;        // 0..7 -> 64-col slice
    const int row0 = blockIdx.x * BROWS;
    const int lm   = lane & 15;
    const int hi   = lane >> 4;

    v8f acc[2][4] = {};

#ifdef USE_TDM
    const uint32_t lds_base = (uint32_t)(uintptr_t)smem;   // flat low 32 = LDS byte offset
    if (wave == 0) {
        tdm_load_2d_bf16(lds_base,                 X + (size_t)row0 * KDIM, KC, BROWS, KDIM);
        tdm_load_2d_bf16(lds_base + BROWS * KC * 2, Wb,                     KC, DIMV,  KDIM);
        __builtin_amdgcn_s_wait_tensorcnt(0);
    }
    __syncthreads();
#endif

    int buf = 0;
    for (int kt = 0; kt < KDIM; kt += KC) {
        const unsigned short* sA = (const unsigned short*)(smem + buf * BUF_B);
        const unsigned short* sB = sA + BROWS * KC;

#ifdef USE_TDM
        // Kick the DMA for the next chunk into the other buffer, then compute.
        if (wave == 0 && kt + KC < KDIM) {
            const uint32_t nb = lds_base + (buf ^ 1) * BUF_B;
            tdm_load_2d_bf16(nb,                 X + (size_t)row0 * KDIM + (kt + KC), KC, BROWS, KDIM);
            tdm_load_2d_bf16(nb + BROWS * KC * 2, Wb + (kt + KC),                     KC, DIMV,  KDIM);
        }
#else
        // Fallback: cooperative global->LDS copy (b128 per thread per step).
        {
            unsigned short* dA = (unsigned short*)(smem + buf * BUF_B);
            unsigned short* dB = dA + BROWS * KC;
            for (int i = threadIdx.x * 8; i < BROWS * KC; i += NTHREADS * 8)
                *(v8u*)(dA + i) = *(const v8u*)(X + (size_t)(row0 + i / KC) * KDIM + kt + (i % KC));
            for (int i = threadIdx.x * 8; i < DIMV * KC; i += NTHREADS * 8)
                *(v8u*)(dB + i) = *(const v8u*)(Wb + (size_t)(i / KC) * KDIM + kt + (i % KC));
            __syncthreads();
        }
#endif

#pragma unroll
        for (int kk = 0; kk < KC / 32; ++kk) {
            // A frag (16x32): lane holds row lm, K = hi*16 + 0..15 contiguous
            // within the 32-wide K-step (same order as B).
            v16bf A[2];
#pragma unroll
            for (int mt = 0; mt < 2; ++mt) {
                const unsigned short* ap = sA + (mt * 16 + lm) * KC + kk * 32 + hi * 16;
                const v16u av = *(const v16u*)ap;
                A[mt] = __builtin_bit_cast(v16bf, av);
            }
            // B frag (32x16): lane holds column d, K = hi*16 + 0..15 contiguous.
            v16bf B[4];
#pragma unroll
            for (int t = 0; t < 4; ++t) {
                const unsigned short* bp =
                    sB + (wave * 64 + t * 16 + lm) * KC + kk * 32 + hi * 16;
                B[t] = __builtin_bit_cast(v16bf, *(const v16u*)bp);
            }
#pragma unroll
            for (int mt = 0; mt < 2; ++mt)
#pragma unroll
                for (int t = 0; t < 4; ++t)
                    acc[mt][t] = __builtin_amdgcn_wmma_f32_16x16x32_bf16(
                        false, A[mt], false, B[t], (short)0, acc[mt][t], false, false);
        }

#ifdef USE_TDM
        if (wave == 0) __builtin_amdgcn_s_wait_tensorcnt(0);  // next buffer landed
#endif
        __syncthreads();
        buf ^= 1;
    }

    // C/D layout: VGPR v, lane L -> M = (L>>4)*8 + v, N = L&15.
#pragma unroll
    for (int t = 0; t < 4; ++t) {
        const int d    = wave * 64 + t * 16 + lm;
        const float bs = pb[d];
#pragma unroll
        for (int mt = 0; mt < 2; ++mt)
#pragma unroll
            for (int v = 0; v < 8; ++v) {
                const int m = row0 + mt * 16 + hi * 8 + v;
                Y[(size_t)m * DIMV + d] = acc[mt][t][v] + bs;
            }
    }
}

// ---------------------------------------------------------------- kernel 4:
// out[b] = dot(Y[b], ent[t[b]])
__global__ __launch_bounds__(NTHREADS) void final_dot_kernel(
    const float* __restrict__ Y, const float* __restrict__ ent,
    const int* __restrict__ t, float* __restrict__ out) {
    __shared__ float red[NTHREADS];
    const int b   = blockIdx.x;
    const int tid = threadIdx.x;
    const float* y = Y + (size_t)b * DIMV;
    const float* e = ent + (size_t)t[b] * DIMV;
    float s = 0.0f;
    for (int d = tid; d < DIMV; d += NTHREADS) s += y[d] * e[d];
    red[tid] = s;
    __syncthreads();
    for (int off = NTHREADS / 2; off > 0; off >>= 1) {
        if (tid < off) red[tid] += red[tid + off];
        __syncthreads();
    }
    if (tid == 0) out[b] = red[0];
}

// ----------------------------------------------------------------
extern "C" void kernel_launch(void* const* d_in, const int* in_sizes, int n_in,
                              void* d_out, int out_size, void* d_ws, size_t ws_size,
                              hipStream_t stream) {
    // setup_inputs order: h, r, t, ent, rel, conv_w, conv_b, proj_w, proj_b
    const int*   h      = (const int*)d_in[0];
    const int*   r      = (const int*)d_in[1];
    const int*   t      = (const int*)d_in[2];
    const float* ent    = (const float*)d_in[3];
    const float* rel    = (const float*)d_in[4];
    const float* conv_w = (const float*)d_in[5];
    const float* conv_b = (const float*)d_in[6];
    const float* proj_w = (const float*)d_in[7];
    const float* proj_b = (const float*)d_in[8];
    float* out = (float*)d_out;

    char* ws = (char*)d_ws;
    unsigned short* Wb = (unsigned short*)ws;                              // 16 MB
    unsigned short* X  = (unsigned short*)(ws + (size_t)DIMV * KDIM * 2);  // 64 MB
    float*          Y  = (float*)(ws + (size_t)DIMV * KDIM * 2
                                     + (size_t)BATCH * KDIM * 2);          // 4 MB
    (void)in_sizes; (void)n_in; (void)out_size; (void)ws_size;

    const int nw4 = DIMV * KDIM / 4;  // 2,097,152 ushort4 groups
    cvt_w_kernel<<<(nw4 + NTHREADS - 1) / NTHREADS, NTHREADS, 0, stream>>>(proj_w, Wb, nw4);
    conv_gather_kernel<<<BATCH, NTHREADS, 0, stream>>>(ent, rel, h, r, conv_w, conv_b, X);
    gemm_wmma_kernel<<<BATCH / BROWS, NTHREADS, 2 * BUF_B, stream>>>(X, Wb, proj_b, Y);
    final_dot_kernel<<<BATCH, NTHREADS, 0, stream>>>(Y, ent, t, out);
}
